// RegisterEncoder_6219112645488
// MI455X (gfx1250) — matrix operand order repaired and checked by reading, not compiled, not run on hardware
//
#include <hip/hip_runtime.h>

// ---------------- common types -----------------------------------------------
typedef __bf16 bf16_t;
typedef __attribute__((ext_vector_type(16))) __bf16 v16bf;
typedef __attribute__((ext_vector_type(8)))  float  v8f;

union BF16Frag { v16bf v; uint4 u[2]; };

__device__ __forceinline__ v8f zero_v8f() {
    v8f z;
#pragma unroll
    for (int i = 0; i < 8; ++i) z[i] = 0.0f;
    return z;
}

#define WMMA_BF16(A_, B_, C_) \
    __builtin_amdgcn_wmma_f32_16x16x32_bf16(false, (A_), false, (B_), (short)0, (C_), false, false)

// Low 32 bits of a shared-aperture flat address are the LDS byte offset
// (ISA 10.2).
__device__ __forceinline__ uint32_t lds32(const void* p) {
    return (uint32_t)(uintptr_t)p;
}

// CDNA5 async DMA: global -> LDS, 16B per lane, tracked by ASYNCcnt.
__device__ __forceinline__ void async_copy_b128(uint32_t lds_dst, const void* gsrc) {
    asm volatile("global_load_async_to_lds_b128 %0, %1, off"
                 :: "v"(lds_dst), "v"(gsrc)
                 : "memory");
}
__device__ __forceinline__ void wait_asynccnt0() {
    asm volatile("s_wait_asynccnt 0x0" ::: "memory");
}
__device__ __forceinline__ void wait_asynccnt4() {   // allow 4 newer in flight
    asm volatile("s_wait_asynccnt 0x4" ::: "memory");
}

// ---------------- problem constants ------------------------------------------
constexpr int kB    = 16;
constexpr int kT    = 512;
constexpr int kNREG = 16;
constexpr int kS    = 528;          // T + NREG
constexpr int kE    = 1024;
constexpr int kNH   = 16;
constexpr int kHD   = 64;
constexpr int kFF   = 4096;
constexpr int kLAT  = 256;
constexpr int kROWS = kB * kS;      // 8448

// ---------------- embed: x = sample@W_emb + b + pos ; concat registers -------
__global__ __launch_bounds__(256) void embed_kernel(
    const float* __restrict__ sample, const float* __restrict__ W_emb,
    const float* __restrict__ b_emb,  const float* __restrict__ pos,
    const float* __restrict__ regs,   float* __restrict__ x)
{
    const int row = blockIdx.x;            // 0 .. B*S-1
    const int b_  = row / kS;
    const int s_  = row % kS;
    const int c   = threadIdx.x * 4;
    float4 r;
    if (s_ < kT) {
        r = *(const float4*)(b_emb + c);
        const float4 pv = *(const float4*)(pos + (size_t)s_ * kE + c);
        r.x += pv.x; r.y += pv.y; r.z += pv.z; r.w += pv.w;
        const float* sp = sample + ((size_t)b_ * kT + s_) * 64;
#pragma unroll 4
        for (int kk = 0; kk < 64; ++kk) {
            const float sv = sp[kk];
            const float4 w = *(const float4*)(W_emb + (size_t)kk * kE + c);
            r.x += sv * w.x; r.y += sv * w.y; r.z += sv * w.z; r.w += sv * w.w;
        }
    } else {
        r = *(const float4*)(regs + (size_t)(s_ - kT) * kE + c);
    }
    *(float4*)(x + (size_t)row * kE + c) = r;
}

// ---------------- LayerNorm (fp32 in -> bf16 out) ----------------------------
__global__ __launch_bounds__(256) void ln_kernel(
    const float* __restrict__ x, bf16_t* __restrict__ out,
    const float* __restrict__ sc, const float* __restrict__ bi, int mode)
{
    const int blk = blockIdx.x;
    const size_t row = (mode == 0) ? (size_t)blk
                                   : (size_t)(blk >> 4) * kS + kT + (blk & 15);
    const float* xr = x + row * kE;
    float v[4], sum = 0.f, sq = 0.f;
#pragma unroll
    for (int j = 0; j < 4; ++j) {
        v[j] = xr[threadIdx.x + 256 * j];
        sum += v[j]; sq += v[j] * v[j];
    }
#pragma unroll
    for (int o = 16; o > 0; o >>= 1) {
        sum += __shfl_xor(sum, o, 32);
        sq  += __shfl_xor(sq,  o, 32);
    }
    __shared__ float s1[8], s2[8];
    if ((threadIdx.x & 31) == 0) { s1[threadIdx.x >> 5] = sum; s2[threadIdx.x >> 5] = sq; }
    __syncthreads();
    float ts = 0.f, tq = 0.f;
#pragma unroll
    for (int i = 0; i < 8; ++i) { ts += s1[i]; tq += s2[i]; }
    const float mean = ts * (1.0f / kE);
    const float var  = tq * (1.0f / kE) - mean * mean;
    const float inv  = rsqrtf(var + 1e-5f);
#pragma unroll
    for (int j = 0; j < 4; ++j) {
        const int e = threadIdx.x + 256 * j;
        out[(size_t)blk * kE + e] = (bf16_t)((v[j] - mean) * inv * sc[e] + bi[e]);
    }
}

// ---------------- fp32 [K,N] -> bf16 [N,K] transpose-convert -----------------
__global__ __launch_bounds__(256) void convtrans_kernel(
    const float* __restrict__ in, bf16_t* __restrict__ out, int K, int N)
{
    __shared__ float tile[32][33];
    const int nb = blockIdx.x, kb = blockIdx.y;
    const int lx = threadIdx.x & 31, ly = threadIdx.x >> 5;
#pragma unroll
    for (int j = 0; j < 4; ++j) {
        const int k = kb * 32 + ly + j * 8, n = nb * 32 + lx;
        tile[ly + j * 8][lx] = in[(size_t)k * N + n];
    }
    __syncthreads();
#pragma unroll
    for (int j = 0; j < 4; ++j) {
        const int n = nb * 32 + ly + j * 8, k = kb * 32 + lx;
        out[(size_t)n * K + k] = (bf16_t)tile[lx][ly + j * 8];
    }
}

// ---------------- WMMA GEMM: C[M,N] = A[M,K] * Bt[N,K]^T (+epilogue) ---------
// Async-DMA double-buffered LDS staging of both 128x32 panels; 8 waves
// (4x2) each compute a 32x64 sub-tile with 8 v_wmma per K=32 step.
enum { EP_QKV = 0, EP_RESID = 1, EP_FC1 = 2, EP_HEAD = 3 };

__device__ __forceinline__ float gelu_tanh(float x) {
    const float c = 0.7978845608028654f;
    return 0.5f * x * (1.0f + tanhf(c * (x + 0.044715f * x * x * x)));
}

constexpr int kLRow = 40;   // LDS row pitch in bf16 elems (80B) -> conflict-free b128 reads

template <int MODE>
__global__ __launch_bounds__(256) void gemm_bf16_kernel(
    const bf16_t* __restrict__ A, const bf16_t* __restrict__ Bt,
    const float* __restrict__ bias,
    float* __restrict__ xres,                 // EP_RESID: residual in/out. EP_HEAD: fp32 out
    bf16_t* __restrict__ outb,                // EP_FC1 out
    bf16_t* __restrict__ outq, bf16_t* __restrict__ outk, bf16_t* __restrict__ outvt,
    int M, int N, int K)
{
    __shared__ alignas(16) bf16_t sA[2][128 * kLRow];
    __shared__ alignas(16) bf16_t sB[2][128 * kLRow];

    const int lane = threadIdx.x & 31;
    const int wave = threadIdx.x >> 5;
    const int wm = wave & 3, wn = wave >> 2;          // 4 x 2 waves -> 128 x 128 tile
    const int half = lane >> 4, l16 = lane & 15;
    const int blkM = blockIdx.y * 128;
    const int blkN = blockIdx.x * 128;
    const int rowbase = blkM + wm * 32;
    const int colbase = blkN + wn * 64;

    // staging: 128 rows x 64B per panel = 512 x 16B chunks; 2 chunks/thread
    const int c0 = threadIdx.x, c1 = threadIdx.x + 256;
    const int r0 = c0 >> 2, i0 = c0 & 3;
    const int r1 = c1 >> 2, i1 = c1 & 3;

    auto stage = [&](int kc, int buf) {
        async_copy_b128(lds32(&sA[buf][r0 * kLRow + i0 * 8]),
                        A + (size_t)(blkM + r0) * K + kc + i0 * 8);
        async_copy_b128(lds32(&sA[buf][r1 * kLRow + i1 * 8]),
                        A + (size_t)(blkM + r1) * K + kc + i1 * 8);
        async_copy_b128(lds32(&sB[buf][r0 * kLRow + i0 * 8]),
                        Bt + (size_t)(blkN + r0) * K + kc + i0 * 8);
        async_copy_b128(lds32(&sB[buf][r1 * kLRow + i1 * 8]),
                        Bt + (size_t)(blkN + r1) * K + kc + i1 * 8);
    };

    v8f acc[2][4];
#pragma unroll
    for (int mi = 0; mi < 2; ++mi)
#pragma unroll
        for (int ni = 0; ni < 4; ++ni) acc[mi][ni] = zero_v8f();

    stage(0, 0);
    const int nk = K >> 5;
    for (int ik = 0; ik < nk; ++ik) {
        const int buf = ik & 1;
        const bool has_next = (ik + 1) < nk;
        if (has_next) {
            stage((ik + 1) << 5, buf ^ 1);
            wait_asynccnt4();                 // older stage complete, newer in flight
        } else {
            wait_asynccnt0();
        }
        __syncthreads();                      // all waves' DMA for `buf` visible

        BF16Frag afr[2], bfr[4];
#pragma unroll
        for (int mi = 0; mi < 2; ++mi) {
            const bf16_t* p = &sA[buf][(wm * 32 + mi * 16 + l16) * kLRow];
            afr[mi].u[0] = *(const uint4*)(p + half * 8);        // K 0..7 / 8..15
            afr[mi].u[1] = *(const uint4*)(p + 16 + half * 8);   // K 16..23 / 24..31
        }
#pragma unroll
        for (int ni = 0; ni < 4; ++ni) {
            const bf16_t* p = &sB[buf][(wn * 64 + ni * 16 + l16) * kLRow] + half * 8;
            bfr[ni].u[0] = *(const uint4*)p;                     // K 0..7 / 8..15
            bfr[ni].u[1] = *(const uint4*)(p + 16);              // K 16..23 / 24..31
        }
#pragma unroll
        for (int mi = 0; mi < 2; ++mi)
#pragma unroll
            for (int ni = 0; ni < 4; ++ni)
                acc[mi][ni] = WMMA_BF16(afr[mi].v, bfr[ni].v, acc[mi][ni]);

        if (has_next) __syncthreads();        // protect buf^1 before next stage
    }

    // epilogue. C layout: N = lane%16, M = vgpr + 8*(lane/16)
#pragma unroll
    for (int mi = 0; mi < 2; ++mi)
#pragma unroll
        for (int ni = 0; ni < 4; ++ni)
#pragma unroll
            for (int vv = 0; vv < 8; ++vv) {
                const int m = rowbase + mi * 16 + vv + half * 8;
                const int n = colbase + ni * 16 + l16;
                float val = acc[mi][ni][vv] + bias[n];
                if constexpr (MODE == EP_QKV) {
                    const int b_ = m / kS, s_ = m % kS;
                    const int which = n >> 10, nn = n & 1023;
                    const int h_ = nn >> 6, d_ = nn & 63;
                    const bf16_t bv = (bf16_t)val;
                    if (which == 0)
                        outq[(((size_t)(b_ * kNH + h_) * kS + s_) << 6) + d_] = bv;
                    else if (which == 1)
                        outk[(((size_t)(b_ * kNH + h_) * kS + s_) << 6) + d_] = bv;
                    else
                        outvt[((size_t)(b_ * kNH + h_) * kHD + d_) * kS + s_] = bv;
                } else if constexpr (MODE == EP_RESID) {
                    const size_t idx = (size_t)m * N + n;
                    xres[idx] += val;
                } else if constexpr (MODE == EP_FC1) {
                    outb[(size_t)m * N + n] = (bf16_t)gelu_tanh(val);
                } else { // EP_HEAD
                    xres[(size_t)m * N + n] = val;
                }
            }
}

// ---------------- flash attention (one 16-row q tile per wave) ---------------
// S^T = K_tile * Q^T (softmax stats per-lane), then O^T = V^T * P with
// cross-half shuffle repack of P into a B-fragment.
__global__ __launch_bounds__(256) void attn_kernel(
    const bf16_t* __restrict__ qg, const bf16_t* __restrict__ kg,
    const bf16_t* __restrict__ vt, bf16_t* __restrict__ o)
{
    const int lane = threadIdx.x & 31;
    const int wave = threadIdx.x >> 5;
    const int task = blockIdx.x * 8 + wave;          // 256 bh * 33 qtiles = 8448
    const int bh = task / 33, qt = task % 33;
    const int half = lane >> 4, l16 = lane & 15;

    BF16Frag qb[2];
    {
        const bf16_t* qrow = qg + ((size_t)bh * kS + qt * 16 + l16) * kHD;
#pragma unroll
        for (int c = 0; c < 2; ++c) {
            const bf16_t* p = qrow + c * 32 + half * 16;
            qb[c].u[0] = *(const uint4*)p;
            qb[c].u[1] = *(const uint4*)(p + 8);
        }
    }

    float m_run = -INFINITY, l_run = 0.0f;
    v8f acc[4];
#pragma unroll
    for (int d = 0; d < 4; ++d) acc[d] = zero_v8f();

    const int qidx = qt * 16 + l16;

    for (int kt = 0; kt < 33; ++kt) {
        BF16Frag ka[2];
        const bf16_t* krow = kg + ((size_t)bh * kS + kt * 16 + l16) * kHD;
#pragma unroll
        for (int c = 0; c < 2; ++c) {
            const bf16_t* p = krow + c * 32 + half * 16;
            ka[c].u[0] = *(const uint4*)p;
            ka[c].u[1] = *(const uint4*)(p + 8);
        }
        v8f ct = zero_v8f();
        ct = WMMA_BF16(ka[0].v, qb[0].v, ct);
        ct = WMMA_BF16(ka[1].v, qb[1].v, ct);

        float p8[8];
        float mloc = -INFINITY;
#pragma unroll
        for (int vv = 0; vv < 8; ++vv) {
            const int kidx = kt * 16 + vv + half * 8;
            float sv = ct[vv] * 0.125f;              // 1/sqrt(64)
            const bool allowed = (qidx < kT) ? (kidx < kT)
                                             : (kidx < kT || kidx <= qidx);
            sv = allowed ? sv : -INFINITY;
            p8[vv] = sv;
            mloc = fmaxf(mloc, sv);
        }
        mloc = fmaxf(mloc, __shfl_xor(mloc, 16, 32));
        const float mnew = fmaxf(m_run, mloc);
        const float corr = __expf(m_run - mnew);
        float psum = 0.0f;
#pragma unroll
        for (int vv = 0; vv < 8; ++vv) {
            p8[vv] = __expf(p8[vv] - mnew);
            psum += p8[vv];
        }
        psum += __shfl_xor(psum, 16, 32);
        l_run = l_run * corr + psum;
        m_run = mnew;
#pragma unroll
        for (int d = 0; d < 4; ++d)
#pragma unroll
            for (int vv = 0; vv < 8; ++vv) acc[d][vv] *= corr;

        float pother[8];
#pragma unroll
        for (int vv = 0; vv < 8; ++vv) pother[vv] = __shfl_xor(p8[vv], 16, 32);
        const float hsel = (half == 0) ? 1.0f : 0.0f;
        BF16Frag pb;
#pragma unroll
        for (int vv = 0; vv < 8; ++vv) {
            pb.v[vv]     = (bf16_t)(p8[vv] * hsel);
            pb.v[vv + 8] = (bf16_t)(pother[vv] * hsel);
        }

#pragma unroll
        for (int dt = 0; dt < 4; ++dt) {
            BF16Frag va;
            const bf16_t* vp = vt + ((size_t)bh * kHD + dt * 16 + l16) * kS
                                  + kt * 16;
            va.u[0] = *(const uint4*)vp;
            va.u[1] = *(const uint4*)(vp + 8);
            acc[dt] = WMMA_BF16(va.v, pb.v, acc[dt]);
        }
    }

    const float invl = 1.0f / l_run;
    const int b_ = bh >> 4, h_ = bh & 15;
    const int s_ = qt * 16 + l16;
#pragma unroll
    for (int dt = 0; dt < 4; ++dt)
#pragma unroll
        for (int vv = 0; vv < 8; ++vv) {
            const int d_ = dt * 16 + vv + half * 8;
            o[((size_t)(b_ * kS + s_)) * kE + h_ * kHD + d_] = (bf16_t)(acc[dt][vv] * invl);
        }
}

// ---------------- host orchestration -----------------------------------------
extern "C" void kernel_launch(void* const* d_in, const int* in_sizes, int n_in,
                              void* d_out, int out_size, void* d_ws, size_t ws_size,
                              hipStream_t stream)
{
    (void)in_sizes; (void)n_in; (void)out_size; (void)ws_size;
    const float* sample = (const float*)d_in[0];
    const float* W_emb  = (const float*)d_in[1];
    const float* b_emb  = (const float*)d_in[2];
    const float* pos    = (const float*)d_in[3];
    const float* regs   = (const float*)d_in[4];
    const float* ln1_s  = (const float*)d_in[5];
    const float* ln1_b  = (const float*)d_in[6];
    const float* W_qkv  = (const float*)d_in[7];
    const float* b_qkv  = (const float*)d_in[8];
    const float* W_o    = (const float*)d_in[9];
    const float* b_o    = (const float*)d_in[10];
    const float* ln2_s  = (const float*)d_in[11];
    const float* ln2_b  = (const float*)d_in[12];
    const float* W_fc1  = (const float*)d_in[13];
    const float* b_fc1  = (const float*)d_in[14];
    const float* W_fc2  = (const float*)d_in[15];
    const float* b_fc2  = (const float*)d_in[16];
    const float* lnf_s  = (const float*)d_in[17];
    const float* lnf_b  = (const float*)d_in[18];
    const float* W_head = (const float*)d_in[19];
    const float* b_head = (const float*)d_in[20];

    char* ws = (char*)d_ws;
    size_t off = 0;
    auto alloc = [&](size_t bytes) -> char* {
        char* p = ws + off;
        off += (bytes + 255) & ~(size_t)255;
        return p;
    };
    float*  x      = (float*) alloc((size_t)kROWS * kE * 4);
    bf16_t* h      = (bf16_t*)alloc((size_t)kROWS * kE * 2);
    bf16_t* qb     = (bf16_t*)alloc((size_t)kROWS * kE * 2);
    bf16_t* kb     = (bf16_t*)alloc((size_t)kROWS * kE * 2);
    bf16_t* vtb    = (bf16_t*)alloc((size_t)kROWS * kE * 2);
    bf16_t* ob     = (bf16_t*)alloc((size_t)kROWS * kE * 2);
    bf16_t* ffb    = (bf16_t*)alloc((size_t)kROWS * kFF * 2);
    bf16_t* hf     = (bf16_t*)alloc((size_t)(kB * kNREG) * kE * 2);
    bf16_t* wt_qkv = (bf16_t*)alloc((size_t)3 * kE * kE * 2);
    bf16_t* wt_o   = (bf16_t*)alloc((size_t)kE * kE * 2);
    bf16_t* wt_f1  = (bf16_t*)alloc((size_t)kFF * kE * 2);
    bf16_t* wt_f2  = (bf16_t*)alloc((size_t)kE * kFF * 2);
    bf16_t* wt_hd  = (bf16_t*)alloc((size_t)kLAT * kE * 2);

    embed_kernel<<<kROWS, 256, 0, stream>>>(sample, W_emb, b_emb, pos, regs, x);

    for (int i = 0; i < 8; ++i) {
        // --- attention block ---
        ln_kernel<<<kROWS, 256, 0, stream>>>(x, h, ln1_s + i * kE, ln1_b + i * kE, 0);
        convtrans_kernel<<<dim3(96, 32), 256, 0, stream>>>(
            W_qkv + (size_t)i * kE * 3 * kE, wt_qkv, kE, 3 * kE);
        gemm_bf16_kernel<EP_QKV><<<dim3(24, 66), 256, 0, stream>>>(
            h, wt_qkv, b_qkv + i * 3 * kE, nullptr, nullptr, qb, kb, vtb,
            kROWS, 3 * kE, kE);
        attn_kernel<<<kROWS / 8, 256, 0, stream>>>(qb, kb, vtb, ob);
        convtrans_kernel<<<dim3(32, 32), 256, 0, stream>>>(
            W_o + (size_t)i * kE * kE, wt_o, kE, kE);
        gemm_bf16_kernel<EP_RESID><<<dim3(8, 66), 256, 0, stream>>>(
            ob, wt_o, b_o + i * kE, x, nullptr, nullptr, nullptr, nullptr,
            kROWS, kE, kE);
        // --- MLP block ---
        ln_kernel<<<kROWS, 256, 0, stream>>>(x, h, ln2_s + i * kE, ln2_b + i * kE, 0);
        convtrans_kernel<<<dim3(128, 32), 256, 0, stream>>>(
            W_fc1 + (size_t)i * kE * kFF, wt_f1, kE, kFF);
        gemm_bf16_kernel<EP_FC1><<<dim3(32, 66), 256, 0, stream>>>(
            h, wt_f1, b_fc1 + i * kFF, nullptr, ffb, nullptr, nullptr, nullptr,
            kROWS, kFF, kE);
        convtrans_kernel<<<dim3(32, 128), 256, 0, stream>>>(
            W_fc2 + (size_t)i * kFF * kE, wt_f2, kFF, kE);
        gemm_bf16_kernel<EP_RESID><<<dim3(8, 66), 256, 0, stream>>>(
            ffb, wt_f2, b_fc2 + i * kE, x, nullptr, nullptr, nullptr, nullptr,
            kROWS, kE, kFF);
    }

    ln_kernel<<<kB * kNREG, 256, 0, stream>>>(x, hf, lnf_s, lnf_b, 1);
    convtrans_kernel<<<dim3(8, 32), 256, 0, stream>>>(W_head, wt_hd, kE, kLAT);
    gemm_bf16_kernel<EP_HEAD><<<dim3(2, 2), 256, 0, stream>>>(
        hf, wt_hd, b_head, (float*)d_out, nullptr, nullptr, nullptr, nullptr,
        kB * kNREG, kLAT, kE);
}
